// Correlation_57612691309207
// MI455X (gfx1250) — matrix operand order repaired and checked by reading, not compiled, not run on hardware
//
#include <hip/hip_runtime.h>
#include <hip/hip_bf16.h>

// FlowNetC correlation on gfx1250 via bf16 WMMA.
// B=8, C=256, H=64, W=96, 21x21 displacements (stride2=2, max_disp=20).

typedef __attribute__((ext_vector_type(16))) __bf16       v16bf;
typedef __attribute__((ext_vector_type(8)))  float        v8f;
typedef __attribute__((ext_vector_type(8)))  unsigned int v8u;

#define B_   8
#define C_   256
#define H_   64
#define W_   96
#define ND   21            // displacements per axis
#define NRA  3             // A x-regions per row
#define NRB  5             // B x-regions per row (with zero guard regions at +-1)
#define TILE_DW 2048       // dwords per packed tile: 16 x * 256 c bf16 = 2048 dw

// ---------------------------------------------------------------------------
// Pack pass: f32 NCHW -> bf16 in per-lane WMMA fragment layout.
// Tile = (b, y, region, parity). Lattice x = xbase + p + 2*idx.
// Tile dword layout: [q(8 K-chunks)][lane(32)][d(8 dwords)] -> lane fragment is
// 32 contiguous bytes => phase-2 loads are two coalesced global_load_b128.
// isA=1: A-matrix (16x32 16-bit) layout, nr=3, scaled by 1/256 (exact pow2).
// isA=0: B-matrix (32x16 16-bit) layout, nr=5 incl. zero guard regions.
// ---------------------------------------------------------------------------
__global__ void pack_kernel(const float* __restrict__ src,
                            unsigned int* __restrict__ dst, int isA, int nr) {
  int g    = blockIdx.x * blockDim.x + threadIdx.x;   // packed dword id
  int d    = g & 7;
  int lane = (g >> 3) & 31;
  int q    = (g >> 8) & 7;
  int tile = g >> 11;
  int p    = tile & 1;
  int rgn  = (tile >> 1) % nr;
  int y    = ((tile >> 1) / nr) % H_;
  int b    = (tile >> 1) / (nr * H_);
  int h    = lane >> 4;      // K-half selector
  int mn   = lane & 15;      // M (A) or N (B) index

  int kk;
  float scale;
  int xbase;
  if (isA) {
    // A 16x32: V0..3 = K{0..7 | 8..15}, V4..7 = K{16..23 | 24..31}
    kk    = ((d >> 2) << 4) + ((d & 3) << 1) + (h << 3);
    scale = 1.0f / 256.0f;   // fold 1/C into A (exact power of two)
    xbase = rgn << 5;
  } else {
    // B 32x16: V0..3 = K{0..7 | 8..15}, V4..7 = K{16..23 | 24..31}
    kk    = ((d >> 2) << 4) + ((d & 3) << 1) + (h << 3);
    scale = 1.0f;
    xbase = (rgn - 1) << 5;  // guard regions at rgn=0 and rgn=4
  }
  int c = (q << 5) + kk;                 // channel of low half (even k)
  int x = xbase + p + (mn << 1);

  unsigned int dw = 0u;
  if (x >= 0 && x < W_) {
    size_t base = ((size_t)(b * C_ + c) * H_ + y) * W_ + x;
    float f0 = src[base] * scale;
    float f1 = src[base + (size_t)H_ * W_] * scale;  // channel c+1
    unsigned short u0 = __builtin_bit_cast(unsigned short, (__bf16)f0);
    unsigned short u1 = __builtin_bit_cast(unsigned short, (__bf16)f1);
    dw = ((unsigned int)u1 << 16) | (unsigned int)u0;
  }
  dst[g] = dw;
}

// ---------------------------------------------------------------------------
// Correlation: one wave per (b, y, region, parity, dy-half). A tile (16
// same-parity x, all 256 ch) in registers; per dy row, 3 B tiles at lattice
// offsets {-32,0,+32} (guard regions make this branch-free) cover all 21 even
// dx shifts exactly once:  D[m,n] with tile t -> dx_idx = n - m + 16t - 6.
// Inner loop: 3 independent accumulator chains interleaved + distance-1
// software pipeline (3 b128-pair loads in flight behind 3 WMMAs).
// ---------------------------------------------------------------------------
__global__ void __launch_bounds__(128)
corr_wmma_kernel(const unsigned int* __restrict__ Apack,
                 const unsigned int* __restrict__ Bpack,
                 float* __restrict__ out) {
  const int tid  = threadIdx.x;
  const int lane = tid & 31;
  const int wid  = tid >> 5;
  const int p    = wid & 1;              // x parity
  const int half = wid >> 1;             // dy interleave: dyi = half, half+2, ...
  const int bid  = blockIdx.x;
  const int r    = bid % 3;
  const int y    = (bid / 3) % H_;
  const int b    = bid / (3 * H_);
  const int h    = lane >> 4;
  const int nn   = lane & 15;

  // ---- load the wave's A tile: 8 chunks x v16bf (2x b128 each, coalesced)
  const unsigned int* at =
      Apack + ((size_t)((b * H_ + y) * NRA + r) * 2 + p) * TILE_DW;
  v16bf afrag[8];
#pragma unroll
  for (int q = 0; q < 8; ++q) {
    v8u av = *(const v8u*)(at + (((q << 5) + lane) << 3));
    afrag[q] = __builtin_bit_cast(v16bf, av);
  }

  // ---- precompute per-lane store offsets/predicates (dy-invariant)
  int offs[3][8];
  int vld[3][8];
#pragma unroll
  for (int t = 0; t < 3; ++t) {
#pragma unroll
    for (int rr = 0; rr < 8; ++rr) {
      int m  = rr + (h << 3);                 // C/D row (M) for this vgpr+lane
      int dx = nn - m + (t << 4) - 6;         // displacement index 0..20
      int xm = (r << 5) + p + (m << 1);       // output x
      offs[t][rr] = dx * (H_ * W_) + xm;
      vld[t][rr]  = (dx >= 0) && (dx < ND);
    }
  }

  float* outby = out + (size_t)b * (ND * ND) * (H_ * W_) + (size_t)y * W_;
  const int laneoff = lane << 3;

  for (int dyi = half; dyi < ND; dyi += 2) {
    const v8f vzero = {0.f, 0.f, 0.f, 0.f, 0.f, 0.f, 0.f, 0.f};
    v8f acc0 = vzero, acc1 = vzero, acc2 = vzero;
    int y2 = y + 2 * dyi - 20;
    if (y2 >= 0 && y2 < H_) {                 // wave-uniform branch
      // base pointers for the 3 B tiles (guard regions => always in-bounds)
      const size_t row = (size_t)(b * H_ + y2) * NRB + r;   // + t in {0,1,2}
      const unsigned int* bt0 = Bpack + ((row + 0) * 2 + p) * TILE_DW + laneoff;
      const unsigned int* bt1 = Bpack + ((row + 1) * 2 + p) * TILE_DW + laneoff;
      const unsigned int* bt2 = Bpack + ((row + 2) * 2 + p) * TILE_DW + laneoff;
      // prologue: stage q=0 fragments for all three tiles
      v8u bs0 = *(const v8u*)(bt0);
      v8u bs1 = *(const v8u*)(bt1);
      v8u bs2 = *(const v8u*)(bt2);
#pragma unroll
      for (int q = 0; q < 8; ++q) {           // K = 256 in 8 steps of 32
        v8u bn0, bn1, bn2;
        if (q < 7) {                           // prefetch next chunk (dist-1)
          const int o = ((q + 1) << 8);
          bn0 = *(const v8u*)(bt0 + o);
          bn1 = *(const v8u*)(bt1 + o);
          bn2 = *(const v8u*)(bt2 + o);
        }
        acc0 = __builtin_amdgcn_wmma_f32_16x16x32_bf16(
                   false, afrag[q], false, __builtin_bit_cast(v16bf, bs0),
                   (short)0, acc0, false, false);
        acc1 = __builtin_amdgcn_wmma_f32_16x16x32_bf16(
                   false, afrag[q], false, __builtin_bit_cast(v16bf, bs1),
                   (short)0, acc1, false, false);
        acc2 = __builtin_amdgcn_wmma_f32_16x16x32_bf16(
                   false, afrag[q], false, __builtin_bit_cast(v16bf, bs2),
                   (short)0, acc2, false, false);
        if (q < 7) { bs0 = bn0; bs1 = bn1; bs2 = bn2; }
      }
    }
    // ---- scatter the 336 useful band entries (zeros for OOB rows)
    float* ob = outby + (size_t)(dyi * ND) * (H_ * W_);
#pragma unroll
    for (int rr = 0; rr < 8; ++rr)
      if (vld[0][rr]) ob[offs[0][rr]] = acc0[rr];
#pragma unroll
    for (int rr = 0; rr < 8; ++rr)
      if (vld[1][rr]) ob[offs[1][rr]] = acc1[rr];
#pragma unroll
    for (int rr = 0; rr < 8; ++rr)
      if (vld[2][rr]) ob[offs[2][rr]] = acc2[rr];
  }
}

// ---------------------------------------------------------------------------
extern "C" void kernel_launch(void* const* d_in, const int* in_sizes, int n_in,
                              void* d_out, int out_size, void* d_ws, size_t ws_size,
                              hipStream_t stream) {
  const float* in1 = (const float*)d_in[0];
  const float* in2 = (const float*)d_in[1];
  float* out = (float*)d_out;

  const size_t apackDw = (size_t)B_ * H_ * NRA * 2 * TILE_DW; //  6,291,456 dw
  const size_t bpackDw = (size_t)B_ * H_ * NRB * 2 * TILE_DW; // 10,485,760 dw
  unsigned int* Apack = (unsigned int*)d_ws;
  unsigned int* Bpack = Apack + apackDw;

  const int threads = 256;
  pack_kernel<<<(int)(apackDw / threads), threads, 0, stream>>>(in1, Apack, 1, NRA);
  pack_kernel<<<(int)(bpackDw / threads), threads, 0, stream>>>(in2, Bpack, 0, NRB);

  // one block = (b, y, region); 4 waves = 2 x-parities x 2 dy-halves
  corr_wmma_kernel<<<B_ * H_ * 3, 128, 0, stream>>>(Apack, Bpack, out);
}
